// SingleWindowAttentionKAN_67714454389361
// MI455X (gfx1250) — compile-verified, matrix operand order checked by reading, not run on hardware
//
#include <hip/hip_runtime.h>

typedef __attribute__((ext_vector_type(16))) __bf16 bf16x16;
typedef __attribute__((ext_vector_type(8)))  __bf16 bf16x8;
typedef __attribute__((ext_vector_type(4)))  __bf16 bf16x4;
typedef __attribute__((ext_vector_type(8)))  float  f32x8;
typedef __attribute__((ext_vector_type(4)))  float  f32x4;

union AFrag { bf16x16 v; bf16x8 h[2]; };

#define MODE_X     0   // A = x (f32 -> bf16)
#define MODE_KCOMB 1   // A = [swish(x) | rbf(x)] fused K = 512 + 2560
#define MODE_BF16  2   // A already bf16 (ctx)

__device__ __forceinline__ float swish_f(float x) {
  return x / (1.0f + __expf(-x));
}

__device__ __forceinline__ bf16x4 pack4(float a, float b, float c, float d) {
  bf16x4 r; r[0] = (__bf16)a; r[1] = (__bf16)b; r[2] = (__bf16)c; r[3] = (__bf16)d;
  return r;
}

// single ds_swizzle_b32 XOR butterfly (group-of-32 mode, and_mask=0x1f).
// masks 8/4/2/1 stay inside each 16-lane half -> width-16 reduction.
#define XSWZ(v, imm) __int_as_float(__builtin_amdgcn_ds_swizzle(__float_as_int(v), imm))
__device__ __forceinline__ float rmax16(float v) {
  v = fmaxf(v, XSWZ(v, 0x201f));   // xor 8
  v = fmaxf(v, XSWZ(v, 0x101f));   // xor 4
  v = fmaxf(v, XSWZ(v, 0x081f));   // xor 2
  v = fmaxf(v, XSWZ(v, 0x041f));   // xor 1
  return v;
}
__device__ __forceinline__ float rsum16(float v) {
  v += XSWZ(v, 0x201f);
  v += XSWZ(v, 0x101f);
  v += XSWZ(v, 0x081f);
  v += XSWZ(v, 0x041f);
  return v;
}

// ---------------------------------------------------------------------------
// Weight pre-pass: Wt[n*ldt + kofs + k] = bf16(W[k*512 + n])  (tiled via LDS)
// ---------------------------------------------------------------------------
__global__ __launch_bounds__(256)
void transpose_w_kernel(const float* __restrict__ W, __bf16* __restrict__ Wt,
                        int ldt, int kofs)
{
  __shared__ float tile[32][33];
  const int tx = threadIdx.x, ty = threadIdx.y;       // block (32, 8)
  const int bk = blockIdx.x * 32, bn = blockIdx.y * 32;
  #pragma unroll
  for (int r = 0; r < 4; ++r)
    tile[ty + 8 * r][tx] = W[(size_t)(bk + ty + 8 * r) * 512 + bn + tx];
  __syncthreads();
  #pragma unroll
  for (int r = 0; r < 4; ++r)
    Wt[(size_t)(bn + ty + 8 * r) * ldt + kofs + bk + tx] =
        (__bf16)tile[tx][ty + 8 * r];
}

// ---------------------------------------------------------------------------
// 128x64-tile bf16 WMMA GEMM: out[M,512] = A(M,K) @ W(K,512) (+bias)
// 256 threads = 8 waves; wave owns a 16-row strip x 64 cols (4 wmma acc).
// Weights pre-transposed bf16 [N][K] -> pure vector staging.
// LDS row stride 40 bf16 (80B): ds_load_b128 gathers 16B-aligned,
// conflict-free across the 16 lanes of a half.
// ---------------------------------------------------------------------------
template <int MODE, bool HAS_BIAS, bool OUT_BF16>
__global__ __launch_bounds__(256)
void kan_gemm_kernel(int K, int Kstride,
                     const float*  __restrict__ xA,    // f32 A source (stride 512)
                     const __bf16* __restrict__ Abf,   // bf16 A source (stride 512)
                     const __bf16* __restrict__ Wt,    // bf16 weights [512][Kstride]
                     const float*  __restrict__ bias,  // bias[512]
                     float*        __restrict__ outF,
                     __bf16*       __restrict__ outB)
{
  const int LDA = 40;
  __shared__ __align__(16) __bf16 As[128 * 40];
  __shared__ __align__(16) __bf16 Bs[64 * 40];

  const int tid  = threadIdx.x;
  const int wave = tid >> 5;
  const int lane = tid & 31;
  const int lh   = lane >> 4;
  const int lm   = lane & 15;
  const int m0   = blockIdx.y * 128;
  const int n0   = blockIdx.x * 64;

  f32x8 acc[4] = {};

  for (int k0 = 0; k0 < K; k0 += 32) {
    // --- stage A tile (128 rows x 32 k): 1024 quads, 4 per thread ---
    #pragma unroll
    for (int j = 0; j < 4; ++j) {
      int i = tid + 256 * j;
      int row = i >> 3, q = i & 7;                  // q: 4-element quad in k
      bf16x4 pk;
      if (MODE == MODE_BF16) {
        pk = *(const bf16x4*)&Abf[(size_t)(m0 + row) * 512 + k0 + 4 * q];
      } else if (MODE == MODE_X) {
        f32x4 v = *(const f32x4*)&xA[(size_t)(m0 + row) * 512 + k0 + 4 * q];
        pk = pack4(v[0], v[1], v[2], v[3]);
      } else { // MODE_KCOMB
        if (k0 < 512) {
          f32x4 v = *(const f32x4*)&xA[(size_t)(m0 + row) * 512 + k0 + 4 * q];
          pk = pack4(swish_f(v[0]), swish_f(v[1]), swish_f(v[2]), swish_f(v[3]));
        } else {
          int kp = k0 - 512 + 4 * q;                // phi col = c*5 + g
          int c = kp / 5, g = kp - 5 * c;           // one div per quad
          float e[4];
          #pragma unroll
          for (int t = 0; t < 4; ++t) {
            float xv = xA[(size_t)(m0 + row) * 512 + c];
            float d  = xv - (-2.0f + (float)g);     // grid step h = 1.0
            e[t] = __expf(-d * d);
            if (++g == 5) { g = 0; ++c; }
          }
          pk = pack4(e[0], e[1], e[2], e[3]);
        }
      }
      *(bf16x4*)&As[row * LDA + 4 * q] = pk;
    }
    // --- stage B tile (64 n x 32 k) from transposed bf16 weights ---
    #pragma unroll
    for (int j = 0; j < 2; ++j) {
      int i = tid + 256 * j;
      int ncol = i >> 3, kq = i & 7;
      *(bf16x4*)&Bs[ncol * LDA + 4 * kq] =
          *(const bf16x4*)&Wt[(size_t)(n0 + ncol) * Kstride + k0 + 4 * kq];
    }
    __syncthreads();

    AFrag a;  // row = 16*wave + lm; K-runs [8*lh,+8) and [16+8*lh,+8)
    a.h[0] = *(const bf16x8*)&As[(16 * wave + lm) * LDA + 8 * lh];
    a.h[1] = *(const bf16x8*)&As[(16 * wave + lm) * LDA + 16 + 8 * lh];
    #pragma unroll
    for (int t = 0; t < 4; ++t) {
      AFrag b;
      b.h[0] = *(const bf16x8*)&Bs[(16 * t + lm) * LDA + 8 * lh];
      b.h[1] = *(const bf16x8*)&Bs[(16 * t + lm) * LDA + 16 + 8 * lh];
      acc[t] = __builtin_amdgcn_wmma_f32_16x16x32_bf16(
          false, a.v, false, b.v, (short)0, acc[t], false, false);
    }
    __syncthreads();
  }

  #pragma unroll
  for (int t = 0; t < 4; ++t) {
    #pragma unroll
    for (int r = 0; r < 8; ++r) {
      int m = m0 + 16 * wave + 8 * lh + r;
      int n = n0 + 16 * t + lm;
      float v = acc[t][r];
      if (HAS_BIAS) v += bias[n];
      if (OUT_BF16) outB[(size_t)m * 512 + n] = (__bf16)v;
      else          outF[(size_t)m * 512 + n] = v;
    }
  }
}

// ---------------------------------------------------------------------------
// Attention: one workgroup (4 waves) per (window b, head h).
// scores 64x64 via 16 WMMAs; scale + rel-pos bias; softmax with single
// ds_swizzle XOR butterflies; ctx = attn @ v via 16 WMMAs.
// ---------------------------------------------------------------------------
__global__ __launch_bounds__(128)
void kan_attn_kernel(const __bf16* __restrict__ qg,
                     const __bf16* __restrict__ kg,
                     const __bf16* __restrict__ vg,
                     const float*  __restrict__ bias_table,  // [225,16]
                     __bf16*       __restrict__ ctxg)        // [32768,512]
{
  __shared__ __align__(16) __bf16 qs [64 * 40];
  __shared__ __align__(16) __bf16 ks [64 * 40];
  __shared__ __align__(16) __bf16 vts[32 * 88];      // v transposed: [j][m]
  __shared__ __align__(16) __bf16 ats[4 * 16 * 88];  // per-wave attn strip

  const int tid  = threadIdx.x;
  const int wave = tid >> 5;
  const int lane = tid & 31;
  const int lh   = lane >> 4;
  const int lm   = lane & 15;
  const int b    = blockIdx.x >> 4;
  const int h    = blockIdx.x & 15;
  const float scale = 0.17677669529663687f;          // 32^-0.5

  // stage q,k,v [64 x 32] bf16 quads (512 quads, 4 per thread)
  #pragma unroll
  for (int j = 0; j < 4; ++j) {
    int i = tid + 128 * j;
    int row = i >> 3, q = i & 7;
    size_t g = ((size_t)(b * 64 + row)) * 512 + h * 32 + 4 * q;
    *(bf16x4*)&qs[row * 40 + 4 * q] = *(const bf16x4*)&qg[g];
    *(bf16x4*)&ks[row * 40 + 4 * q] = *(const bf16x4*)&kg[g];
    bf16x4 vv = *(const bf16x4*)&vg[g];
    #pragma unroll
    for (int c = 0; c < 4; ++c) vts[(4 * q + c) * 88 + row] = vv[c];
  }
  __syncthreads();

  // scores strip: rows [16*wave, 16*wave+16), K = hd = 32
  f32x8 acc[4];
  {
    AFrag a;
    a.h[0] = *(const bf16x8*)&qs[(16 * wave + lm) * 40 + 8 * lh];
    a.h[1] = *(const bf16x8*)&qs[(16 * wave + lm) * 40 + 16 + 8 * lh];
    #pragma unroll
    for (int t = 0; t < 4; ++t) {
      AFrag bb;  // B = k^T: column n of B is k row n
      bb.h[0] = *(const bf16x8*)&ks[(16 * t + lm) * 40 + 8 * lh];
      bb.h[1] = *(const bf16x8*)&ks[(16 * t + lm) * 40 + 16 + 8 * lh];
      f32x8 z = {};
      acc[t] = __builtin_amdgcn_wmma_f32_16x16x32_bf16(
          false, a.v, false, bb.v, (short)0, z, false, false);
    }
  }

  // scale + relative-position bias, row softmax (row m lives in vgpr m%8
  // across one 16-lane half, 4 n-tiles deep)
  #pragma unroll
  for (int r = 0; r < 8; ++r) {
    int m = 16 * wave + 8 * lh + r;
    float s[4];
    float mx = -3.0e38f;
    #pragma unroll
    for (int t = 0; t < 4; ++t) {
      int n = 16 * t + lm;
      int idx = ((m >> 3) - (n >> 3) + 7) * 15 + ((m & 7) - (n & 7) + 7);
      s[t] = acc[t][r] * scale + bias_table[idx * 16 + h];
      mx = fmaxf(mx, s[t]);
    }
    mx = rmax16(mx);
    float sum = 0.0f;
    #pragma unroll
    for (int t = 0; t < 4; ++t) { s[t] = __expf(s[t] - mx); sum += s[t]; }
    sum = rsum16(sum);
    float inv = 1.0f / sum;
    #pragma unroll
    for (int t = 0; t < 4; ++t) acc[t][r] = s[t] * inv;
  }

  // re-stage attn strip as bf16 in WMMA A layout
  #pragma unroll
  for (int t = 0; t < 4; ++t)
    #pragma unroll
    for (int r = 0; r < 8; ++r)
      ats[wave * (16 * 88) + (8 * lh + r) * 88 + 16 * t + lm] = (__bf16)acc[t][r];
  __syncthreads();

  // ctx strip = attn[16x64] @ v[64x32]; two 32-wide K steps
  f32x8 c2[2] = {};
  #pragma unroll
  for (int s = 0; s < 2; ++s) {
    AFrag a;
    const __bf16* ab = &ats[wave * (16 * 88) + lm * 88 + 32 * s + 8 * lh];
    a.h[0] = *(const bf16x8*)ab;
    a.h[1] = *(const bf16x8*)(ab + 16);
    #pragma unroll
    for (int tn = 0; tn < 2; ++tn) {
      AFrag bb;  // B = v: column j of B is vts row j
      const __bf16* vb = &vts[(16 * tn + lm) * 88 + 32 * s + 8 * lh];
      bb.h[0] = *(const bf16x8*)vb;
      bb.h[1] = *(const bf16x8*)(vb + 16);
      c2[tn] = __builtin_amdgcn_wmma_f32_16x16x32_bf16(
          false, a.v, false, bb.v, (short)0, c2[tn], false, false);
    }
  }

  // ctx straight into [B*N, C] layout for the projection GEMM
  #pragma unroll
  for (int tn = 0; tn < 2; ++tn)
    #pragma unroll
    for (int r = 0; r < 8; ++r) {
      int row = b * 64 + 16 * wave + 8 * lh + r;
      int col = h * 32 + 16 * tn + lm;
      ctxg[(size_t)row * 512 + col] = (__bf16)c2[tn][r];
    }
}

extern "C" void kernel_launch(void* const* d_in, const int* in_sizes, int n_in,
                              void* d_out, int out_size, void* d_ws, size_t ws_size,
                              hipStream_t stream) {
  (void)in_sizes; (void)n_in; (void)out_size; (void)ws_size;
  const float* x         = (const float*)d_in[0];
  const float* Wq        = (const float*)d_in[1];
  const float* Wk_base   = (const float*)d_in[2];
  const float* Wk_spline = (const float*)d_in[3];
  const float* Wv        = (const float*)d_in[4];
  const float* Wproj     = (const float*)d_in[5];
  const float* bproj     = (const float*)d_in[6];
  const float* bias_tab  = (const float*)d_in[7];

  const size_t ROWS = 512ull * 64ull;   // 32768
  const size_t ELT  = ROWS * 512ull;    // 16,777,216
  __bf16* qg   = (__bf16*)d_ws;         // 32 MB each
  __bf16* kg   = qg + ELT;
  __bf16* vg   = kg + ELT;
  __bf16* ctxg = vg + ELT;
  __bf16* Wqt  = ctxg + ELT;            // [512][512]
  __bf16* Wvt  = Wqt + 512 * 512;
  __bf16* Wpt  = Wvt + 512 * 512;
  __bf16* Wkt  = Wpt + 512 * 512;       // [512][3072]: base | spline

  dim3 tblk(32, 8);
  transpose_w_kernel<<<dim3(16, 16), tblk, 0, stream>>>(Wq,        Wqt,  512,   0);
  transpose_w_kernel<<<dim3(16, 16), tblk, 0, stream>>>(Wv,        Wvt,  512,   0);
  transpose_w_kernel<<<dim3(16, 16), tblk, 0, stream>>>(Wproj,     Wpt,  512,   0);
  transpose_w_kernel<<<dim3(16, 16), tblk, 0, stream>>>(Wk_base,   Wkt, 3072,   0);
  transpose_w_kernel<<<dim3(80, 16), tblk, 0, stream>>>(Wk_spline, Wkt, 3072, 512);

  dim3 blk(256);
  dim3 ggrid(512 / 64, (unsigned)(ROWS / 128));   // (8, 256)

  // Q = x @ Wq
  kan_gemm_kernel<MODE_X, false, true><<<ggrid, blk, 0, stream>>>(
      512, 512, x, nullptr, Wqt, nullptr, nullptr, qg);
  // K = swish(x) @ Wk_base + rbf(x) @ Wk_spline  (fused K = 3072)
  kan_gemm_kernel<MODE_KCOMB, false, true><<<ggrid, blk, 0, stream>>>(
      3072, 3072, x, nullptr, Wkt, nullptr, nullptr, kg);
  // V = x @ Wv
  kan_gemm_kernel<MODE_X, false, true><<<ggrid, blk, 0, stream>>>(
      512, 512, x, nullptr, Wvt, nullptr, nullptr, vg);
  // attention per (window, head)
  kan_attn_kernel<<<dim3(512 * 16), dim3(128), 0, stream>>>(qg, kg, vg,
                                                            bias_tab, ctxg);
  // out = ctx @ Wproj + bproj (f32)
  kan_gemm_kernel<MODE_BF16, true, false><<<ggrid, blk, 0, stream>>>(
      512, 512, nullptr, ctxg, Wpt, bproj, (float*)d_out, nullptr);
}
